// minGRU2_50130858279604
// MI455X (gfx1250) — compile-verified, matrix operand order checked by reading, not run on hardware
//
#include <hip/hip_runtime.h>
#include <hip/hip_bf16.h>

// ---------------------------------------------------------------------------
// minGRU bidirectional, MI455X (gfx1250, wave32)
//   B=8, D=512, L=8192.  out (8,512,8192) f32.
// Pipeline:
//   1) norm_w_kernel:    Wn = normed_weight(W) -> bf16  (1024x512)
//   2) transpose_kernel: xT[b][l][k] = (bf16) x[b][k][l]   (B,L,D) bf16
//   3) gemm_kernel:      fb[b] = Wn @ x[b]  via v_wmma_f32_16x16x32_bf16,
//                        async global->LDS tile copies, double-buffered
//   4) scan_kernel:      linear recurrence h_t = a_t h_{t-1} + b_t, fwd+bwd
// Workspace layout (d_ws):
//   [0,        1 MiB)  Wn  bf16
//   [1 MiB,   65 MiB)  xT  bf16
//   [65 MiB, 321 MiB)  fb  f32
// ---------------------------------------------------------------------------

typedef __attribute__((ext_vector_type(16))) __bf16 v16bf;
typedef __attribute__((ext_vector_type(8)))  __bf16 v8bf;
typedef __attribute__((ext_vector_type(8)))  float  v8f;

union FragBF16 {
    v16bf v;
    v8bf  h[2];
};

#define NB 8
#define ND 512
#define NL 8192
#define NM 1024          // 2*D output rows of the GEMM

// ---------------------------------------------------------------------------
// Async global->LDS copy (CDNA5 GLOBAL_LOAD_ASYNC_TO_LDS_B128, ASYNCcnt).
// The builtin takes (int4 addrspace(1)*, int4 addrspace(3)*, imm, imm).
// Guarded: falls back to sync load+ds_store if the builtin is absent.
// ---------------------------------------------------------------------------
#if __has_builtin(__builtin_amdgcn_global_load_async_to_lds_b128)
#define HAVE_ASYNC_LDS 1
#else
#define HAVE_ASYNC_LDS 0
#endif

typedef int v4i32 __attribute__((vector_size(16)));
typedef __attribute__((address_space(1))) v4i32 gv4i;
typedef __attribute__((address_space(3))) v4i32 sv4i;

__device__ __forceinline__ void copy16_to_lds(const __bf16* g, __bf16* l)
{
#if HAVE_ASYNC_LDS
    __builtin_amdgcn_global_load_async_to_lds_b128((gv4i*)g, (sv4i*)l, 0, 0);
#else
    *(v8bf*)l = *(const v8bf*)g;
#endif
}

__device__ __forceinline__ void wait_async_copies()
{
#if HAVE_ASYNC_LDS
#if __has_builtin(__builtin_amdgcn_s_wait_asynccnt)
    __builtin_amdgcn_s_wait_asynccnt(0);
#else
    asm volatile("s_wait_asynccnt 0" ::: "memory");
#endif
#endif
}

// ---------------------------------------------------------------------------
// 1) Per-row weight normalization.  One block per output row (1024 blocks).
// ---------------------------------------------------------------------------
__global__ __launch_bounds__(256)
void norm_w_kernel(const float* __restrict__ W, __bf16* __restrict__ Wn)
{
    const int o = blockIdx.x;
    const float* row = W + (size_t)o * ND;

    float s = 0.f;
    for (int i = threadIdx.x; i < ND; i += 256) {
        float w = row[i];
        s += w * w;
    }
    for (int off = 16; off > 0; off >>= 1)
        s += __shfl_down(s, off, 32);

    __shared__ float partial[8];
    if ((threadIdx.x & 31) == 0) partial[threadIdx.x >> 5] = s;
    __syncthreads();
    if (threadIdx.x == 0) {
        float t = 0.f;
        for (int i = 0; i < 8; ++i) t += partial[i];
        partial[0] = t;
    }
    __syncthreads();

    const float inv_sqrt_fanin = 0.04419417382415922f;   // 1/sqrt(512)
    const float sqrt_fanin     = 22.62741699796952f;     // sqrt(512)
    float norm = sqrtf(partial[0]);
    float inv  = 1.f / ((1e-4f + norm * inv_sqrt_fanin) * sqrt_fanin);

    for (int i = threadIdx.x; i < ND; i += 256)
        Wn[(size_t)o * ND + i] = (__bf16)(row[i] * inv);
}

// ---------------------------------------------------------------------------
// 2) Transpose+convert: x (B,D,L) f32 -> xT (B,L,D) bf16, tiled via LDS.
// ---------------------------------------------------------------------------
__global__ __launch_bounds__(256)
void transpose_kernel(const float* __restrict__ x, __bf16* __restrict__ xT)
{
    __shared__ float tile[32][33];
    const int l0 = blockIdx.x * 32;
    const int d0 = blockIdx.y * 32;
    const int b  = blockIdx.z;
    const int tx = threadIdx.x;      // 0..31
    const int ty = threadIdx.y;      // 0..7

    const float* xb = x + (size_t)b * ND * NL;
    for (int i = 0; i < 32; i += 8)
        tile[ty + i][tx] = xb[(size_t)(d0 + ty + i) * NL + l0 + tx];
    __syncthreads();

    __bf16* xTb = xT + (size_t)b * NL * ND;
    for (int i = 0; i < 32; i += 8)
        xTb[(size_t)(l0 + ty + i) * ND + d0 + tx] = (__bf16)tile[tx][ty + i];
}

// ---------------------------------------------------------------------------
// 3) GEMM: fb[b] = Wn(1024x512) @ x[b](512x8192), bf16 inputs, f32 accum.
//    Block tile 128(M) x 128(N), BK=32. 8 waves; wave tile 32x64 = 2x4 WMMA.
//    Tiles staged to LDS with async copies, double-buffered.
// ---------------------------------------------------------------------------
#define BM 128
#define BN 128
#define BK 32
#define LDSK 40   // padded row (80 B): keeps 16-B alignment, skews banks

__global__ __launch_bounds__(256)
void gemm_kernel(const __bf16* __restrict__ Wn, const __bf16* __restrict__ xT,
                 float* __restrict__ fb)
{
    __shared__ __bf16 sA[2][BM][LDSK];
    __shared__ __bf16 sB[2][BN][LDSK];

    const int tid  = threadIdx.x;
    const int lane = tid & 31;
    const int wave = tid >> 5;
    const int half = lane >> 4;      // 0: lanes 0-15, 1: lanes 16-31
    const int l15  = lane & 15;

    const int n0 = blockIdx.x * BN;
    const int m0 = blockIdx.y * BM;
    const int b  = blockIdx.z;

    const int wm = wave >> 1;        // 0..3  (M direction, 32 rows each)
    const int wn = wave & 1;         // 0..1  (N direction, 64 cols each)

    const __bf16* xb = xT + (size_t)b * NL * ND;

    // Issue one tile's async copies: 512 chunks of 16 B per matrix, 2/thread.
    auto issue_tile = [&](int buf, int k0) {
        for (int i = 0; i < 2; ++i) {
            const int chunk = tid + i * 256;   // 0..511
            const int row   = chunk >> 2;      // 0..127
            const int kc    = chunk & 3;       // 0..3 (8 bf16 each)
            copy16_to_lds(Wn + (size_t)(m0 + row) * ND + k0 + kc * 8,
                          &sA[buf][row][kc * 8]);
            copy16_to_lds(xb + (size_t)(n0 + row) * ND + k0 + kc * 8,
                          &sB[buf][row][kc * 8]);
        }
    };

    v8f acc[2][4];
    for (int mt = 0; mt < 2; ++mt)
        for (int nt = 0; nt < 4; ++nt)
            acc[mt][nt] = (v8f){0.f, 0.f, 0.f, 0.f, 0.f, 0.f, 0.f, 0.f};

    const int NK = ND / BK;          // 16 K-steps
    issue_tile(0, 0);

    for (int ki = 0; ki < NK; ++ki) {
        const int cur = ki & 1;

        // Our copies for buf[cur] have landed; barrier makes everyone's
        // visible and guarantees all waves are done reading buf[cur^1].
        wait_async_copies();
        __syncthreads();

        if (ki + 1 < NK)
            issue_tile(cur ^ 1, (ki + 1) * BK);

        // Fragments per the documented 16-bit A/B layout:
        //   low 8 elems  = K[half*8 .. half*8+7]
        //   high 8 elems = K[16+half*8 .. 16+half*8+7]
        FragBF16 afr[2], bfr[4];
        for (int mt = 0; mt < 2; ++mt) {
            const int r = wm * 32 + mt * 16 + l15;
            afr[mt].h[0] = *(const v8bf*)&sA[cur][r][half * 8];
            afr[mt].h[1] = *(const v8bf*)&sA[cur][r][16 + half * 8];
        }
        for (int nt = 0; nt < 4; ++nt) {
            const int r = wn * 64 + nt * 16 + l15;
            bfr[nt].h[0] = *(const v8bf*)&sB[cur][r][half * 8];
            bfr[nt].h[1] = *(const v8bf*)&sB[cur][r][16 + half * 8];
        }

        for (int mt = 0; mt < 2; ++mt)
            for (int nt = 0; nt < 4; ++nt)
                acc[mt][nt] = __builtin_amdgcn_wmma_f32_16x16x32_bf16(
                    false, afr[mt].v, false, bfr[nt].v,
                    (short)0, acc[mt][nt], false, false);
    }

    // Writeback: C 16x16 f32 layout — VGPR j: M = half*8 + j, N = lane&15.
    for (int mt = 0; mt < 2; ++mt) {
        for (int nt = 0; nt < 4; ++nt) {
            const int n = n0 + wn * 64 + nt * 16 + l15;
            for (int j = 0; j < 8; ++j) {
                const int m = m0 + wm * 32 + mt * 16 + half * 8 + j;
                fb[((size_t)b * NM + m) * NL + n] = acc[mt][nt][j];
            }
        }
    }
}

// ---------------------------------------------------------------------------
// 4) minGRU scan.  One wave per (batch, channel, direction) sequence.
//    Each lane owns a 256-step chunk; affine maps combined across lanes
//    with a log-step shfl scan; second pass emits outputs (L2-resident
//    re-read: each wave re-touches its own 64 KB immediately).
// ---------------------------------------------------------------------------
__device__ __forceinline__ void mingru_coef(float g, float h, float& a, float& bv)
{
    // a = sigmoid(-g)*scale, b = sigmoid(g)*scale*sign(h)*max(|h|,1e-6)
    // scale = sqrt(1/cosh(g) + 1), all evaluated overflow-safely.
    float eng      = __expf(-fabsf(g));
    float inv_cosh = 2.f * eng / (1.f + eng * eng);
    float scale    = sqrtf(1.f + inv_cosh);
    float sig      = 1.f / (1.f + __expf(-g));
    float hs       = copysignf(fmaxf(fabsf(h), 1e-6f), h);
    a  = (1.f - sig) * scale;
    bv = sig * scale * hs;
}

__global__ __launch_bounds__(256)
void scan_kernel(const float* __restrict__ fb, float* __restrict__ out)
{
    const int wid  = (int)((blockIdx.x * blockDim.x + threadIdx.x) >> 5);
    const int lane = threadIdx.x & 31;
    if (wid >= NB * 256 * 2) return;

    const int dir = wid & 1;            // 0 = forward, 1 = backward
    const int c   = (wid >> 1) & 255;
    const int b   = wid >> 9;

    const float* hrow = fb  + ((size_t)b * NM  + (dir ? 512 + c : c))       * NL;
    const float* grow = fb  + ((size_t)b * NM  + (dir ? 768 + c : 256 + c)) * NL;
    float*       orow = out + ((size_t)b * 512 + (dir ? 256 + c : c))       * NL;

    const int CHUNK = NL / 32;          // 256
    const int t0 = lane * CHUNK;

    // Pass 1: per-lane affine composition over its chunk.
    float A = 1.f, Bv = 0.f;
    for (int i = 0; i < CHUNK; ++i) {
        const int t   = t0 + i;
        const int idx = dir ? (NL - 1 - t) : t;
        float a, bv;
        mingru_coef(grow[idx], hrow[idx], a, bv);
        A  = a * A;
        Bv = a * Bv + bv;
    }

    // Inclusive scan of affine maps across lanes (lane 0 earliest).
    for (int off = 1; off < 32; off <<= 1) {
        float Ap = __shfl_up(A,  off, 32);
        float Bp = __shfl_up(Bv, off, 32);
        if (lane >= off) {
            Bv = A * Bp + Bv;
            A  = A * Ap;
        }
    }
    float carry = __shfl_up(Bv, 1, 32);
    if (lane == 0) carry = 0.f;

    // Pass 2: replay chunk with carried-in state, emit outputs.
    float hst = carry;
    for (int i = 0; i < CHUNK; ++i) {
        const int t   = t0 + i;
        const int idx = dir ? (NL - 1 - t) : t;
        float a, bv;
        mingru_coef(grow[idx], hrow[idx], a, bv);
        hst = a * hst + bv;
        orow[idx] = hst;
    }
}

// ---------------------------------------------------------------------------
extern "C" void kernel_launch(void* const* d_in, const int* in_sizes, int n_in,
                              void* d_out, int out_size, void* d_ws, size_t ws_size,
                              hipStream_t stream)
{
    const float* x = (const float*)d_in[0];   // (8, 512, 8192) f32
    const float* W = (const float*)d_in[1];   // (1024, 512, 1) f32
    float*     out = (float*)d_out;           // (8, 512, 8192) f32

    char* ws = (char*)d_ws;
    __bf16* Wn = (__bf16*)ws;                                  //   1 MiB
    __bf16* xT = (__bf16*)(ws + ((size_t)1 << 20));            //  64 MiB
    float*  fb = (float*)(ws + ((size_t)65 << 20));            // 256 MiB

    norm_w_kernel<<<NM, 256, 0, stream>>>(W, Wn);
    transpose_kernel<<<dim3(NL / 32, ND / 32, NB), dim3(32, 8), 0, stream>>>(x, xT);
    gemm_kernel<<<dim3(NL / BN, NM / BM, NB), 256, 0, stream>>>(Wn, xT, fb);
    scan_kernel<<<(NB * 256 * 2 * 32) / 256, 256, 0, stream>>>(fb, out);
}
